// RelationConditionedEntityAttention_14680198218393
// MI455X (gfx1250) — matrix operand order repaired and checked by reading, not compiled, not run on hardware
//
#include <hip/hip_runtime.h>

#define Bdim 4
#define Ndim 256
#define Hdim 128
#define Rdim 16

typedef _Float16 f16;
typedef __attribute__((ext_vector_type(16))) _Float16 v16h;
typedef __attribute__((ext_vector_type(8)))  _Float16 v8h;
typedef __attribute__((ext_vector_type(8)))  float    v8f;
typedef __attribute__((ext_vector_type(4)))  float    v4f;

static __device__ __forceinline__ v16h cat8(v8h lo, v8h hi) {
  return __builtin_shufflevector(lo, hi, 0,1,2,3,4,5,6,7,8,9,10,11,12,13,14,15);
}
static __device__ __forceinline__ v8h ld8h(const f16* p) { return *(const v8h*)p; }
static __device__ __forceinline__ v4f ld4f(const float* p){ return *(const v4f*)p; }

// A-layout fragment (16x32 MxK, 16-bit): lane row = lane&15,
// element e -> K = 16g+e (g = lane>=16)
static __device__ __forceinline__ v16h fragA(const f16* rowp, int s, int g) {
  const f16* p = rowp + 32*s + 16*g;
  return cat8(ld8h(p), ld8h(p + 8));
}
// B-layout fragment (32x16 KxN, 16-bit): lane col = lane&15, element e -> K = 16g+e
static __device__ __forceinline__ v16h fragB(const f16* rowp, int s, int g) {
  const f16* p = rowp + 32*s + 16*g;
  return cat8(ld8h(p), ld8h(p + 8));
}
static __device__ __forceinline__ v8f wmma16(v16h a, v16h b, v8f c) {
  return __builtin_amdgcn_wmma_f32_16x16x32_f16(false, a, false, b, (short)0, c, false, false);
}

// CDNA5 async global->LDS copy (ASYNCcnt-tracked, no VGPR round trip)
static __device__ __forceinline__ unsigned lds_off(void* p) {
  return (unsigned)(size_t)p;   // low 32 bits of generic shared pointer = LDS address
}
static __device__ __forceinline__ void async_b128(unsigned ldsaddr, const void* g) {
  asm volatile("global_load_async_to_lds_b128 %0, %1, off"
               :: "v"(ldsaddr), "v"((unsigned long long)(size_t)g) : "memory");
}
static __device__ __forceinline__ void wait_async() {
  asm volatile("s_wait_asynccnt 0x0" ::: "memory");
}

// ---------------- prep kernels ----------------
__global__ __launch_bounds__(256)
void prep_weights(const float* __restrict__ Wq, const float* __restrict__ Wk,
                  const float* __restrict__ Wv, const float* __restrict__ W1,
                  const float* __restrict__ W2, const float* __restrict__ Gamma,
                  const float* __restrict__ Beta,
                  f16* Wqh, f16* Wkh, f16* Wvh, f16* W1ah, f16* W1bh, f16* W1ch,
                  f16* W2h, f16* GT2, f16* BT2, float* zbias)
{
  int idx = blockIdx.x * 256 + threadIdx.x;
  int sel = blockIdx.y;
  if (sel < 3) {
    if (idx < Hdim*Hdim) {
      const float* s = sel==0?Wq : sel==1?Wk : Wv;
      f16* d = sel==0?Wqh : sel==1?Wkh : Wvh;
      d[idx] = (f16)s[idx];
    }
  } else if (sel < 6) {
    if (idx < Hdim*Hdim) {
      int k = idx >> 7, h = idx & 127;
      f16* d = sel==3?W1ah : sel==4?W1bh : W1ch;
      d[idx] = (f16)W1[k*(3*Hdim) + (sel-3)*Hdim + h];
    }
  } else if (sel == 6) {
    if (idx < Rdim*Hdim) W2h[idx] = (f16)W2[idx];
  } else if (sel < 9) {
    // Gamma^T / Beta^T padded to 32 rows of r so WMMA B-frags are branchless
    if (idx < Hdim*32) {
      int h = idx >> 5, r = idx & 31;
      f16* d = sel==7 ? GT2 : BT2;
      const float* s = sel==7 ? Gamma : Beta;
      d[idx] = (r < Rdim) ? (f16)s[r*Hdim + h] : (f16)0.f;
    }
  } else {
    if (idx < Hdim) zbias[idx] = 0.f;
  }
}

__global__ __launch_bounds__(256)
void prep_E(const float* __restrict__ embs, const unsigned char* __restrict__ mask,
            float* __restrict__ Ef32, f16* __restrict__ Ef16)
{
  int idx = blockIdx.x * 256 + threadIdx.x;
  if (idx >= Bdim*Ndim*Hdim) return;
  int row = idx >> 7;
  float v = mask[row] ? embs[idx] : 0.f;
  Ef32[idx] = v; Ef16[idx] = (f16)v;
}

// ---------------- projection kernels (branchless, one output each) ----------------
__global__ __launch_bounds__(32)
void proj_f16(const f16* __restrict__ Ef16, const f16* __restrict__ W,
              const float* __restrict__ bias, f16* __restrict__ outp)
{
  int lane = threadIdx.x & 31, l15 = lane & 15, g = lane >> 4;
  int rowbase = blockIdx.x * 16;
  const f16* arow = Ef16 + (size_t)(rowbase + l15)*Hdim;
  #pragma unroll
  for (int nt = 0; nt < 8; ++nt) {
    v8f acc = {};
    #pragma unroll
    for (int s = 0; s < 4; ++s)
      acc = wmma16(fragA(arow, s, g), fragB(W + (size_t)(16*nt + l15)*Hdim, s, g), acc);
    int col = 16*nt + l15;
    float bb = bias[col];
    #pragma unroll
    for (int k = 0; k < 8; ++k)
      outp[(size_t)(rowbase + k + 8*g)*Hdim + col] = (f16)(acc[k] + bb);
  }
}

__global__ __launch_bounds__(32)
void proj_f32(const f16* __restrict__ Ef16, const f16* __restrict__ W,
              const float* __restrict__ bias, float* __restrict__ outp)
{
  int lane = threadIdx.x & 31, l15 = lane & 15, g = lane >> 4;
  int rowbase = blockIdx.x * 16;
  const f16* arow = Ef16 + (size_t)(rowbase + l15)*Hdim;
  #pragma unroll
  for (int nt = 0; nt < 8; ++nt) {
    v8f acc = {};
    #pragma unroll
    for (int s = 0; s < 4; ++s)
      acc = wmma16(fragA(arow, s, g), fragB(W + (size_t)(16*nt + l15)*Hdim, s, g), acc);
    int col = 16*nt + l15;
    float bb = bias[col];
    #pragma unroll
    for (int k = 0; k < 8; ++k)
      outp[(size_t)(rowbase + k + 8*g)*Hdim + col] = acc[k] + bb;
  }
}

// ---------------- fused main kernel ----------------
#define WAVE_SZ   37056
#define SHARED_SZ 20480
#define LDS_TOTAL (SHARED_SZ + 4*WAVE_SZ)
// wave region: Ej f32 [0,8192) | Vj f32 [8192,16384) | HJ f32 [16384,24576)
//              msc f32 [24576,32768) | Kj f16 [32768,36864)
//              Mrun [36864) Srun [36928) Pex [36992)

__global__ __launch_bounds__(128)
void rcea_main(const float* __restrict__ Ef32, const f16* __restrict__ Qh,
               const f16* __restrict__ Kh, const float* __restrict__ Vf,
               const float* __restrict__ HIg, const float* __restrict__ HJg,
               const f16* __restrict__ W1ch, const f16* __restrict__ W2h,
               const f16* __restrict__ GT2, const f16* __restrict__ BT2,
               const float* __restrict__ b1, const float* __restrict__ b2,
               const float* __restrict__ relb, const float* __restrict__ lnw,
               const float* __restrict__ lnb, const unsigned char* __restrict__ mask,
               float* __restrict__ out)
{
  extern __shared__ char smem[];
  const int tid = threadIdx.x;
  const int wave = tid >> 5, lane = tid & 31, l15 = lane & 15, g = lane >> 4;
  const int b = blockIdx.x >> 4, it = blockIdx.x & 15;
  const int ibase = it * 16;

  float* Ei  = (float*)(smem);
  float* HIi = (float*)(smem + 8192);
  f16*   Qi  = (f16*)(smem + 16384);

  char* wb = smem + SHARED_SZ + wave*WAVE_SZ;
  float* Ej   = (float*)(wb);
  float* Vj   = (float*)(wb + 8192);
  float* HJ   = (float*)(wb + 16384);
  float* msc  = (float*)(wb + 24576);
  f16*   Kj   = (f16*)(wb + 32768);
  float* Mrun = (float*)(wb + 36864);
  float* Srun = (float*)(wb + 36928);
  float* Pex  = (float*)(wb + 36992);

  { // async i-tile staging (Ei, HIi f32; Qi f16)
    const char* sE = (const char*)(Ef32 + (size_t)(b*Ndim + ibase)*Hdim);
    const char* sH = (const char*)(HIg  + (size_t)(b*Ndim + ibase)*Hdim);
    const char* sQ = (const char*)(Qh   + (size_t)(b*Ndim + ibase)*Hdim);
    unsigned dE = lds_off(Ei), dH = lds_off(HIi), dQ = lds_off(Qi);
    #pragma unroll
    for (int t = 0; t < 4; ++t) {
      unsigned byt = (tid + 128*t) * 16u;
      async_b128(dE + byt, sE + byt);
      async_b128(dH + byt, sH + byt);
    }
    #pragma unroll
    for (int t = 0; t < 2; ++t) {
      unsigned byt = (tid + 128*t) * 16u;
      async_b128(dQ + byt, sQ + byt);
    }
  }
  { // per-wave accumulator init
    v4f z = {0.f,0.f,0.f,0.f};
    v4f* m4 = (v4f*)msc;
    #pragma unroll
    for (int t = 0; t < 16; ++t) m4[lane + 32*t] = z;
    if (lane < 16) { Mrun[lane] = -3.0e38f; Srun[lane] = 0.f; }
  }
  wait_async();
  __syncthreads();

  float b2v[8], rbv[8];
  #pragma unroll
  for (int k = 0; k < 8; ++k) { b2v[k] = b2[8*g + k]; rbv[k] = relb[8*g + k]; }

  for (int jt = wave; jt < 16; jt += 4) {
    const int jbase = jt * 16;
    { // async j-tile staging into wave-private LDS
      const char* sE = (const char*)(Ef32 + (size_t)(b*Ndim + jbase)*Hdim);
      const char* sV = (const char*)(Vf   + (size_t)(b*Ndim + jbase)*Hdim);
      const char* sH = (const char*)(HJg  + (size_t)(b*Ndim + jbase)*Hdim);
      const char* sK = (const char*)(Kh   + (size_t)(b*Ndim + jbase)*Hdim);
      unsigned dE = lds_off(Ej), dV = lds_off(Vj), dH = lds_off(HJ), dK = lds_off(Kj);
      #pragma unroll
      for (int t = 0; t < 16; ++t) {
        unsigned byt = (lane + 32*t) * 16u;
        async_b128(dE + byt, sE + byt);
        async_b128(dV + byt, sV + byt);
        async_b128(dH + byt, sH + byt);
      }
      #pragma unroll
      for (int t = 0; t < 8; ++t) {
        unsigned byt = (lane + 32*t) * 16u;
        async_b128(dK + byt, sK + byt);
      }
      wait_async();
    }
    float mpen = mask[b*Ndim + jbase + l15] ? 0.f : -1e9f;

    // scores tile: q_i . k_j  (C-layout: lane col j=l15, rows i)
    v8f sv = {};
    #pragma unroll
    for (int s = 0; s < 4; ++s)
      sv = wmma16(fragA(Qi + l15*Hdim, s, g), fragB(Kj + l15*Hdim, s, g), sv);

    for (int m = 0; m < 16; ++m) {
      // B-fragments of P^T[h,j] = E_i[m][h]*E_j[j][h]
      v16h pb[4];
      #pragma unroll
      for (int s = 0; s < 4; ++s) {
        const float* pe = Ei + m*Hdim + 32*s + 16*g;
        const float* pj = Ej + l15*Hdim + 32*s + 16*g;
        v16h t;
        #pragma unroll
        for (int q4 = 0; q4 < 4; ++q4) {
          v4f a = ld4f(pe + 4*q4), c = ld4f(pj + 4*q4);
          #pragma unroll
          for (int k = 0; k < 4; ++k) t[4*q4 + k] = (f16)(a[k]*c[k]);
        }
        pb[s] = t;
      }
      // GEMM1: hiddenT[k_out, j] = W1c @ P^T  (+hi +hj +b1, relu)
      v8f acc[8];
      #pragma unroll
      for (int mt = 0; mt < 8; ++mt) {
        v8f a = {};
        #pragma unroll
        for (int s = 0; s < 4; ++s)
          a = wmma16(fragA(W1ch + (size_t)(16*mt + l15)*Hdim, s, g), pb[s], a);
        const float* hi = HIi + m*Hdim + mt*16 + 8*g;
        const float* hj = HJ  + l15*Hdim + mt*16 + 8*g;
        const float* bb = b1  + mt*16 + 8*g;
        v4f h0=ld4f(hi), h1=ld4f(hi+4), j0=ld4f(hj), j1=ld4f(hj+4), c0=ld4f(bb), c1=ld4f(bb+4);
        #pragma unroll
        for (int k = 0; k < 4; ++k) {
          a[k]   = fmaxf(a[k]   + h0[k] + j0[k] + c0[k], 0.f);
          a[k+4] = fmaxf(a[k+4] + h1[k] + j1[k] + c1[k], 0.f);
        }
        acc[mt] = a;
      }
      // GEMM2: rel_logits^T[r,j] = W2 @ hiddenT  (B-frags via xor16 shuffles)
      v8f d2 = {};
      #pragma unroll
      for (int s2 = 0; s2 < 4; ++s2) {
        v16h bh;
        #pragma unroll
        for (int k = 0; k < 8; ++k) {
          float lo = acc[2*s2][k], hi2 = acc[2*s2+1][k];
          float tmp = g ? lo : hi2;
          float sw = __shfl_xor(tmp, 16, 32);
          bh[k]   = (f16)(g ? sw : lo);
          bh[k+8] = (f16)(g ? hi2 : sw);
        }
        d2 = wmma16(fragA(W2h + (size_t)l15*Hdim, s2, g), bh, d2);
      }
      // softmax over R (rows of d2: within-lane 8 + partner 8)
      float pr[8]; float rmax = -3.0e38f;
      #pragma unroll
      for (int k = 0; k < 8; ++k) { pr[k] = d2[k] + b2v[k]; rmax = fmaxf(rmax, pr[k]); }
      rmax = fmaxf(rmax, __shfl_xor(rmax, 16, 32));
      float rsum = 0.f;
      #pragma unroll
      for (int k = 0; k < 8; ++k) { pr[k] = __expf(pr[k]-rmax); rsum += pr[k]; }
      rsum += __shfl_xor(rsum, 16, 32);
      float rinv = 1.f / rsum, rb = 0.f;
      #pragma unroll
      for (int k = 0; k < 8; ++k) { pr[k] *= rinv; rb += pr[k]*rbv[k]; }
      rb += __shfl_xor(rb, 16, 32);
      // score(m, j=l15) + online softmax over the 16 columns
      float tq = (g == (m >> 3)) ? sv[m & 7] : 0.f;
      float sc = (tq + __shfl_xor(tq, 16, 32)) * 0.08838834764831845f + rb + mpen;
      float tmax = sc;
      #pragma unroll
      for (int d = 1; d < 16; d <<= 1) tmax = fmaxf(tmax, __shfl_xor(tmax, d, 32));
      float Mold = Mrun[m];
      float Mnew = fmaxf(Mold, tmax);
      float scale = __expf(Mold - Mnew);
      float pe = __expf(sc - Mnew);
      float psum = pe;
      #pragma unroll
      for (int d = 1; d < 16; d <<= 1) psum += __shfl_xor(psum, d, 32);
      // duplicate lanes write identical values: benign LDS conflicts
      Srun[m] = Srun[m]*scale + psum;
      Mrun[m] = Mnew;
      #pragma unroll
      for (int k = 0; k < 4; ++k) { int ix = m*Hdim + lane + 32*k; msc[ix] *= scale; }
      Pex[l15] = pe;
      v4f px0 = ld4f(Pex + 8*g), px1 = ld4f(Pex + 8*g + 4);
      float pex[8] = {px0[0],px0[1],px0[2],px0[3],px1[0],px1[1],px1[2],px1[3]};
      // probs A-fragment (K padded 16->32 with zeros)
      v16h a3;
      #pragma unroll
      for (int k = 0; k < 8; ++k) { a3[k] = (f16)pr[k]; a3[k+8] = (f16)0.f; }
      // GEMM3: gamma/beta = probs @ Gamma^T/Beta^T (branchless, zero-padded B)
      #pragma unroll
      for (int nt = 0; nt < 8; ++nt) {
        const f16* gp = GT2 + (size_t)(16*nt + l15)*32 + 8*g;
        const f16* bp = BT2 + (size_t)(16*nt + l15)*32 + 8*g;
        v16h bg  = cat8(ld8h(gp), ld8h(gp + 16));
        v16h bbv = cat8(ld8h(bp), ld8h(bp + 16));
        v8f zero = {};
        v8f dg = wmma16(a3, bg, zero);
        v8f db = wmma16(a3, bbv, zero);
        float part = 0.f;
        #pragma unroll
        for (int k = 0; k < 8; ++k) {
          float vv = Vj[(k + 8*g)*Hdim + 16*nt + l15];
          part += pex[k] * (vv * (1.f + dg[k]) + db[k]);
        }
        atomicAdd(&msc[m*Hdim + 16*nt + l15], part);  // ds_add_f32, both half-waves
      }
    } // m
  } // jt
  __syncthreads();

  // merge 4 waves' online-softmax states + LayerNorm
  for (int mm = wave; mm < 16; mm += 4) {
    float Mg = -3.0e38f;
    #pragma unroll
    for (int w = 0; w < 4; ++w)
      Mg = fmaxf(Mg, ((float*)(smem + SHARED_SZ + w*WAVE_SZ + 36864))[mm]);
    float Sg = 0.f, fac[4];
    #pragma unroll
    for (int w = 0; w < 4; ++w) {
      char* o = smem + SHARED_SZ + w*WAVE_SZ;
      fac[w] = __expf(((float*)(o + 36864))[mm] - Mg);
      Sg += ((float*)(o + 36928))[mm] * fac[w];
    }
    float inv = 1.f / Sg;
    int row = b*Ndim + ibase + mm;
    float xv[4];
    #pragma unroll
    for (int k = 0; k < 4; ++k) {
      int h = lane + 32*k;
      float msg = 0.f;
      #pragma unroll
      for (int w = 0; w < 4; ++w)
        msg += ((float*)(smem + SHARED_SZ + w*WAVE_SZ + 24576))[mm*Hdim + h] * fac[w];
      xv[k] = Ef32[(size_t)row*Hdim + h] + msg*inv;
    }
    float s = xv[0]+xv[1]+xv[2]+xv[3];
    #pragma unroll
    for (int d = 1; d < 32; d <<= 1) s += __shfl_xor(s, d, 32);
    float mu = s * (1.f/128.f);
    float vs = 0.f;
    #pragma unroll
    for (int k = 0; k < 4; ++k) { float dd = xv[k]-mu; vs += dd*dd; }
    #pragma unroll
    for (int d = 1; d < 32; d <<= 1) vs += __shfl_xor(vs, d, 32);
    float rstd = rsqrtf(vs*(1.f/128.f) + 1e-5f);
    float rowm = mask[row] ? 1.f : 0.f;
    #pragma unroll
    for (int k = 0; k < 4; ++k) {
      int h = lane + 32*k;
      out[(size_t)row*Hdim + h] = ((xv[k]-mu)*rstd*lnw[h] + lnb[h]) * rowm;
    }
  }
}

// ---------------- launcher ----------------
extern "C" void kernel_launch(void* const* d_in, const int* in_sizes, int n_in,
                              void* d_out, int out_size, void* d_ws, size_t ws_size,
                              hipStream_t stream) {
  const float* embs = (const float*)d_in[0];
  const unsigned char* mask = (const unsigned char*)d_in[1];
  const float* Wq = (const float*)d_in[2];  const float* bq = (const float*)d_in[3];
  const float* Wk = (const float*)d_in[4];  const float* bk = (const float*)d_in[5];
  const float* Wv = (const float*)d_in[6];  const float* bv = (const float*)d_in[7];
  const float* W1 = (const float*)d_in[8];  const float* b1 = (const float*)d_in[9];
  const float* W2 = (const float*)d_in[10]; const float* b2 = (const float*)d_in[11];
  const float* Gamma = (const float*)d_in[12];
  const float* Beta  = (const float*)d_in[13];
  const float* relb  = (const float*)d_in[14];
  const float* lnw   = (const float*)d_in[15];
  const float* lnb   = (const float*)d_in[16];
  float* out = (float*)d_out;

  char* w = (char*)d_ws;
  size_t o = 0;
  auto carve = [&](size_t bytes) { char* p = w + o; o += (bytes + 255) & ~(size_t)255; return p; };
  float* Ef32 = (float*)carve(Bdim*Ndim*Hdim*4);
  f16* Ef16   = (f16*)carve(Bdim*Ndim*Hdim*2);
  f16* Qh     = (f16*)carve(Bdim*Ndim*Hdim*2);
  f16* Kh     = (f16*)carve(Bdim*Ndim*Hdim*2);
  float* Vf   = (float*)carve(Bdim*Ndim*Hdim*4);
  float* HIf  = (float*)carve(Bdim*Ndim*Hdim*4);
  float* HJf  = (float*)carve(Bdim*Ndim*Hdim*4);
  f16* Wqh  = (f16*)carve(Hdim*Hdim*2);
  f16* Wkh  = (f16*)carve(Hdim*Hdim*2);
  f16* Wvh  = (f16*)carve(Hdim*Hdim*2);
  f16* W1ah = (f16*)carve(Hdim*Hdim*2);
  f16* W1bh = (f16*)carve(Hdim*Hdim*2);
  f16* W1ch = (f16*)carve(Hdim*Hdim*2);
  f16* W2h  = (f16*)carve(Rdim*Hdim*2);
  f16* GT2  = (f16*)carve(Hdim*32*2);
  f16* BT2  = (f16*)carve(Hdim*32*2);
  float* zb = (float*)carve(Hdim*4);
  (void)in_sizes; (void)n_in; (void)out_size; (void)ws_size;

  prep_weights<<<dim3(64, 10), 256, 0, stream>>>(Wq, Wk, Wv, W1, W2, Gamma, Beta,
      Wqh, Wkh, Wvh, W1ah, W1bh, W1ch, W2h, GT2, BT2, zb);
  prep_E<<<dim3((Bdim*Ndim*Hdim + 255)/256), 256, 0, stream>>>(embs, mask, Ef32, Ef16);
  proj_f16<<<dim3(Bdim*Ndim/16), 32, 0, stream>>>(Ef16, Wqh, bq, Qh);
  proj_f16<<<dim3(Bdim*Ndim/16), 32, 0, stream>>>(Ef16, Wkh, bk, Kh);
  proj_f32<<<dim3(Bdim*Ndim/16), 32, 0, stream>>>(Ef16, Wvh, bv, Vf);
  proj_f32<<<dim3(Bdim*Ndim/16), 32, 0, stream>>>(Ef16, W1ah, zb, HIf);
  proj_f32<<<dim3(Bdim*Ndim/16), 32, 0, stream>>>(Ef16, W1bh, zb, HJf);
  rcea_main<<<dim3(Bdim*16), 128, LDS_TOTAL, stream>>>(Ef32, Qh, Kh, Vf, HIf, HJf,
      W1ch, W2h, GT2, BT2, b1, b2, relb, lnw, lnb, mask, out);
}
